// EdgeLevelRNN_73151882986080
// MI455X (gfx1250) — compile-verified, matrix operand order checked by reading, not run on hardware
//
#include <hip/hip_runtime.h>

// ---------------------------------------------------------------------------
// EdgeLevelRNN for MI455X (gfx1250): wave32, v_wmma_f32_16x16x32_f16 GEMMs,
// TDM (tensor_load_to_lds) double-buffered weight staging where available.
// ---------------------------------------------------------------------------

typedef __attribute__((ext_vector_type(16))) _Float16 v16h;
typedef __attribute__((ext_vector_type(8)))  float    v8f;

#define Bdim 1024
#define Tdim 64
#define Ndim 128
#define Edim 256
#define Hdim 256
#define HEADS 4
#define DH 64

// ---- feature probes -------------------------------------------------------
#if defined(__HIP_DEVICE_COMPILE__) &&                                    \
    __has_builtin(__builtin_amdgcn_tensor_load_to_lds) &&                 \
    __has_builtin(__builtin_amdgcn_s_wait_tensorcnt)
#define USE_TDM 1
#else
#define USE_TDM 0
#endif

#if __has_include(<hip/amd_detail/amd_gfx1250_TDM.h>)
#define TDM_6ARG 1   // therock-10.0 headers -> 6-arg builtin
#else
#define TDM_6ARG 0   // ROCm 7.2 -> 5-arg builtin
#endif

// ---- WMMA fragment helper -------------------------------------------------
// 16-bit A/B fragment layout (CDNA5 ISA 7.12.2):
//   lane 0-15  : row M (A) / col N (B), K pairs: regs0-3 -> K 0..7, regs4-7 -> K 16..23
//   lane 16-31 : same row,              K pairs: regs0-3 -> K 8..15, regs4-7 -> K 24..31
__device__ __forceinline__ v16h frag_load(const _Float16* p, int half) {
  v16h f;
#pragma unroll
  for (int i = 0; i < 8; ++i) {
    int k = ((i & 4) ? 16 : 0) + half * 8 + ((i & 3) << 1);
    f[2 * i]     = p[k];
    f[2 * i + 1] = p[k + 1];
  }
  return f;
}

__device__ __forceinline__ v8f wmma16(v16h a, v16h b, v8f c) {
  return __builtin_amdgcn_wmma_f32_16x16x32_f16(false, a, false, b, (short)0, c,
                                                false, false);
}

#if USE_TDM
typedef unsigned int tdm_u32x4 __attribute__((ext_vector_type(4)));
typedef int          tdm_i32x4 __attribute__((ext_vector_type(4)));
typedef int          tdm_i32x8 __attribute__((ext_vector_type(8)));

// TDM load of one 128-row x 32-half W tile into LDS with row padding 32->40
// halves (pad_interval = 16 DWORDs, pad_amount = 4 DWORDs).  D# layout per
// CDNA5 ISA.
__device__ __forceinline__ void tdm_load_w_tile(const _Float16* gsrc,
                                                unsigned lds_byte_off,
                                                int Ktotal) {
  unsigned long long ga = (unsigned long long)(uintptr_t)gsrc;
  tdm_u32x4 g0;
  g0[0] = 1u;                                              // count=1 (valid)
  g0[1] = lds_byte_off;                                    // lds_addr
  g0[2] = (unsigned)ga;                                    // global_addr[31:0]
  g0[3] = (unsigned)((ga >> 32) & 0x01ffffffu) | (2u << 30);  // [56:32] | type=2
  tdm_i32x8 g1;
  g1[0] = (1 << 16)        // data_size = 2 bytes
        | (1 << 20)        // pad_enable
        | (3 << 22)        // pad_interval: 16 DWORDs (64B) between pads
        | (3 << 25);       // pad_amount: 4 DWORDs (8 halves)
  g1[1] = (Ktotal & 0xffff) << 16;                         // tensor_dim0[15:0]
  g1[2] = ((Ktotal >> 16) & 0xffff) | (128 << 16);         // dim0 hi | tensor_dim1 lo
  g1[3] = (32 << 16);                                      // dim1 hi=0 | tile_dim0=32
  g1[4] = 128;                                             // tile_dim1=128, tile_dim2=0
  g1[5] = Ktotal;                                          // tensor_dim0_stride lo
  g1[6] = 0;
  g1[7] = 0;
  tdm_i32x4 z4 = {0, 0, 0, 0};
#if TDM_6ARG
  tdm_i32x8 z8 = {0, 0, 0, 0, 0, 0, 0, 0};
  __builtin_amdgcn_tensor_load_to_lds(g0, g1, z4, z4, z8, 0);
#else
  __builtin_amdgcn_tensor_load_to_lds(g0, g1, z4, z4, 0);
#endif
}
#endif  // USE_TDM

// ---- Generic tiled WMMA GEMM:  C = act(A * W^T + bias) --------------------
// A: M x K (row-major, lda), W: N x K (row-major, f16)
// Outputs: optional f32 C (ldcf) and/or f16 C (ldch).  act: 0=none,1=relu,2=*0.125
// vtrans=1: scatter f16 output as (B, 256, 128) "v transposed" layout.
template <typename AT>
__global__ __launch_bounds__(256) void gemm_kernel(
    const AT* __restrict__ A, int lda,
    const _Float16* __restrict__ W,
    const float* __restrict__ bias,
    float* __restrict__ Cf, int ldcf,
    _Float16* __restrict__ Ch, int ldch,
    int M, int N, int K, int act, int vtrans) {
  __shared__ _Float16 As[32][40];
  __shared__ _Float16 Ws[2][128][40];

  const int tid  = threadIdx.x;
  const int lane = tid & 31, wid = tid >> 5;
  const int half = lane >> 4, r = lane & 15;
  const int wm = wid >> 2, wn = wid & 3;          // 2 x 4 wave grid
  const int m0 = blockIdx.x * 32;
  const int n0 = blockIdx.y * 128;
  const int nk = K >> 5;

  v8f acc0 = {};
  v8f acc1 = {};

#if USE_TDM
  const bool issuer = (wid == 0);
  if (issuer)  // preload first W tile via the tensor data mover
    tdm_load_w_tile(&W[(size_t)n0 * K],
                    (unsigned)(uintptr_t)(void*)&Ws[0][0][0], K);
#endif

  for (int kc = 0; kc < nk; ++kc) {
    const int k0 = kc << 5;
    const int buf = kc & 1;

#if USE_TDM
    if (issuer) {
      if (kc + 1 < nk) {   // issue next tile, then wait current (in-order)
        tdm_load_w_tile(&W[(size_t)n0 * K + k0 + 32],
                        (unsigned)(uintptr_t)(void*)&Ws[buf ^ 1][0][0], K);
        __builtin_amdgcn_s_wait_tensorcnt(1);
      } else {
        __builtin_amdgcn_s_wait_tensorcnt(0);
      }
    }
#else
    // Stage W tile (128x32) into LDS with 16B vector copies.
    {
      int col = tid >> 1;
      int kk  = (tid & 1) * 16;
      const uint4* s4 = (const uint4*)&W[(size_t)(n0 + col) * K + k0 + kk];
      uint4* d4 = (uint4*)&Ws[buf][col][kk];
      d4[0] = s4[0];
      d4[1] = s4[1];
    }
#endif
    // Stage A tile (32x32) into LDS, converting to f16 when needed.
    {
      int e   = tid * 4;                 // 4 elements per thread
      int row = e >> 5, kk = e & 31;
      const AT* src = &A[(size_t)(m0 + row) * lda + k0 + kk];
      _Float16* dst = &As[row][kk];
      if constexpr (sizeof(AT) == 2) {
        *(uint2*)dst = *(const uint2*)src;
      } else {
#pragma unroll
        for (int j = 0; j < 4; ++j) dst[j] = (_Float16)(float)src[j];
      }
    }
    __syncthreads();

    v16h a  = frag_load(&As[wm * 16 + r][0], half);
    v16h b0 = frag_load(&Ws[buf][wn * 32 + r][0], half);
    v16h b1 = frag_load(&Ws[buf][wn * 32 + 16 + r][0], half);
    acc0 = wmma16(a, b0, acc0);
    acc1 = wmma16(a, b1, acc1);
    __syncthreads();
  }

  // Epilogue.  C/D layout: lanes0-15 N=lane, reg g -> M=g; lanes16-31 -> M=8+g.
#pragma unroll
  for (int ct = 0; ct < 2; ++ct) {
    v8f acc = ct ? acc1 : acc0;
#pragma unroll
    for (int g = 0; g < 8; ++g) {
      int row = m0 + wm * 16 + g + 8 * half;
      int col = n0 + wn * 32 + ct * 16 + r;
      float v = acc[g];
      if (bias) v += bias[col];
      if (act == 1) v = fmaxf(v, 0.0f);
      else if (act == 2) v *= 0.125f;   // 1/sqrt(DH)
      if (Cf) Cf[(size_t)row * ldcf + col] = v;
      if (Ch) {
        if (vtrans)
          Ch[((size_t)(row >> 7) * 256 + col) * 128 + (row & 127)] = (_Float16)v;
        else
          Ch[(size_t)row * ldch + col] = (_Float16)v;
      }
    }
  }
}

// ---- Attention: one wave per (batch, head) --------------------------------
// scores(1x128) = q(1x64) K^T ; softmax ; ctx(1x64) = a(1x128) V
// Q / softmax weights broadcast into all 16 A rows so row 0 of D is valid in
// every lane's reg0.
__global__ __launch_bounds__(256) void attention_kernel(
    const _Float16* __restrict__ q,     // (B, 256)
    const _Float16* __restrict__ kmat,  // (B, N, 256)
    const _Float16* __restrict__ vt,    // (B, 256, N)  transposed v
    const unsigned char* __restrict__ mask,  // (B, N)
    _Float16* __restrict__ ctx) {       // (B, 256)
  __shared__ _Float16 a_lds[8][128];

  const int lane = threadIdx.x & 31, wid = threadIdx.x >> 5;
  const int half = lane >> 4, r = lane & 15;
  const int task = blockIdx.x * 8 + wid;   // B*HEADS tasks
  const int b = task >> 2, head = task & 3;

  const _Float16* qp = &q[(size_t)b * 256 + head * 64];
  v16h qa0 = frag_load(qp, half);        // K 0..31  (broadcast rows)
  v16h qa1 = frag_load(qp + 32, half);   // K 32..63

  float s[8];
#pragma unroll
  for (int ct = 0; ct < 8; ++ct) {
    const _Float16* kp = &kmat[((size_t)b * 128 + ct * 16 + r) * 256 + head * 64];
    v16h kb0 = frag_load(kp, half);
    v16h kb1 = frag_load(kp + 32, half);
    v8f acc = {};
    acc = wmma16(qa0, kb0, acc);
    acc = wmma16(qa1, kb1, acc);
    s[ct] = acc[0];                      // row0, col = ct*16 + r
  }

  // mask + softmax over 128 (halves hold duplicates: width-16 reductions)
  float mx = -3.0e38f;
#pragma unroll
  for (int ct = 0; ct < 8; ++ct) {
    int n = ct * 16 + r;
    if (mask[(size_t)b * 128 + n]) s[ct] = -1.0e9f;
    mx = fmaxf(mx, s[ct]);
  }
#pragma unroll
  for (int off = 8; off >= 1; off >>= 1) mx = fmaxf(mx, __shfl_xor(mx, off, 16));
  float e[8], sum = 0.0f;
#pragma unroll
  for (int ct = 0; ct < 8; ++ct) {
    e[ct] = __expf(s[ct] - mx);
    sum += e[ct];
  }
#pragma unroll
  for (int off = 8; off >= 1; off >>= 1) sum += __shfl_xor(sum, off, 16);
  float inv = 1.0f / sum;

  if (lane < 16) {
#pragma unroll
    for (int ct = 0; ct < 8; ++ct)
      a_lds[wid][ct * 16 + lane] = (_Float16)(e[ct] * inv);
  }
  __syncthreads();

  v16h af[4];
#pragma unroll
  for (int c = 0; c < 4; ++c) af[c] = frag_load(&a_lds[wid][c * 32], half);

#pragma unroll
  for (int dt = 0; dt < 4; ++dt) {
    const _Float16* vp = &vt[((size_t)b * 256 + head * 64 + dt * 16 + r) * 128];
    v8f acc = {};
#pragma unroll
    for (int c = 0; c < 4; ++c) {
      v16h vb = frag_load(vp + c * 32, half);
      acc = wmma16(af[c], vb, acc);
    }
    if (lane < 16)
      ctx[(size_t)b * 256 + head * 64 + dt * 16 + lane] = (_Float16)acc[0];
  }
}

// ---- Pointwise kernels ----------------------------------------------------
__global__ void f32_to_f16_kernel(const float* __restrict__ s,
                                  _Float16* __restrict__ d, int n) {
  int i = blockIdx.x * 256 + threadIdx.x;
  if (i < n) d[i] = (_Float16)s[i];
}

__global__ void init_h_kernel(const float* __restrict__ h0in,
                              float* __restrict__ h0f, _Float16* __restrict__ h0h,
                              float* __restrict__ h1f, _Float16* __restrict__ cat) {
  int idx = blockIdx.x * 256 + threadIdx.x;  // B*H
  int b = idx >> 8, j = idx & 255;
  float a = h0in[idx];
  float c = h0in[Bdim * Hdim + idx];
  h0f[idx] = a;
  h0h[idx] = (_Float16)a;
  h1f[idx] = c;
  cat[(size_t)b * 512 + j] = (_Float16)c;
}

__global__ void emb_step_kernel(const float* __restrict__ x,
                                const float* __restrict__ W_in,
                                const float* __restrict__ b_in,
                                _Float16* __restrict__ emb, int t) {
  int idx = blockIdx.x * 256 + threadIdx.x;  // B*E
  int b = idx >> 8, e = idx & 255;
  float v = fmaxf(0.0f, x[(size_t)b * Tdim + t] * W_in[e] + b_in[e]);
  emb[idx] = (_Float16)v;
}

__global__ void gru_pw_kernel(const float* __restrict__ gi,
                              const float* __restrict__ gh,
                              const float* __restrict__ bih,
                              const float* __restrict__ bhh,
                              const float* __restrict__ hprev,
                              float* __restrict__ hnew,
                              _Float16* __restrict__ hnew_h, int ldh) {
  int idx = blockIdx.x * 256 + threadIdx.x;  // B*H
  int b = idx >> 8, j = idx & 255;
  size_t g = (size_t)b * 768;
  float ir = gi[g + j] + bih[j],              hr = gh[g + j] + bhh[j];
  float iz = gi[g + 256 + j] + bih[256 + j],  hz = gh[g + 256 + j] + bhh[256 + j];
  float in_ = gi[g + 512 + j] + bih[512 + j], hn = gh[g + 512 + j] + bhh[512 + j];
  float rr = 1.0f / (1.0f + __expf(-(ir + hr)));
  float zz = 1.0f / (1.0f + __expf(-(iz + hz)));
  float nn = tanhf(in_ + rr * hn);
  float h = (1.0f - zz) * nn + zz * hprev[idx];
  hnew[idx] = h;
  hnew_h[(size_t)b * ldh + j] = (_Float16)h;
}

// logits = out1 . W2 + b2 ; sigmoid ; write (B,T) slot t.  One wave per row.
__global__ void final_out_kernel(const _Float16* __restrict__ out1,
                                 const _Float16* __restrict__ W2h,
                                 const float* __restrict__ b2,
                                 float* __restrict__ out, int t) {
  int wid = threadIdx.x >> 5, lane = threadIdx.x & 31;
  int b = blockIdx.x * 8 + wid;
  const _Float16* p = &out1[(size_t)b * 256];
  float s = 0.0f;
#pragma unroll
  for (int i = 0; i < 8; ++i)
    s += (float)p[lane + i * 32] * (float)W2h[lane + i * 32];
#pragma unroll
  for (int off = 16; off >= 1; off >>= 1) s += __shfl_xor(s, off, 32);
  if (lane == 0)
    out[(size_t)b * Tdim + t] = 1.0f / (1.0f + __expf(-(s + b2[0])));
}

// ---------------------------------------------------------------------------
extern "C" void kernel_launch(void* const* d_in, const int* in_sizes, int n_in,
                              void* d_out, int out_size, void* d_ws, size_t ws_size,
                              hipStream_t stream) {
  (void)in_sizes; (void)n_in; (void)out_size; (void)ws_size;
  const float* x    = (const float*)d_in[0];
  const float* prev = (const float*)d_in[1];
  const float* h0in = (const float*)d_in[2];
  const float* W_in = (const float*)d_in[3];
  const float* b_in = (const float*)d_in[4];
  const float* Wih0 = (const float*)d_in[5];
  const float* Whh0 = (const float*)d_in[6];
  const float* bih0 = (const float*)d_in[7];
  const float* bhh0 = (const float*)d_in[8];
  const float* Wih1 = (const float*)d_in[9];
  const float* Whh1 = (const float*)d_in[10];
  const float* bih1 = (const float*)d_in[11];
  const float* bhh1 = (const float*)d_in[12];
  const float* Wq = (const float*)d_in[13];
  const float* Wk = (const float*)d_in[14];
  const float* Wv = (const float*)d_in[15];
  const float* bq = (const float*)d_in[16];
  const float* bk = (const float*)d_in[17];
  const float* bv = (const float*)d_in[18];
  const float* Wo = (const float*)d_in[19];
  const float* bo = (const float*)d_in[20];
  const float* Wc = (const float*)d_in[21];
  const float* bc = (const float*)d_in[22];
  const float* W1 = (const float*)d_in[23];
  const float* b1 = (const float*)d_in[24];
  const float* W2 = (const float*)d_in[25];
  const float* b2 = (const float*)d_in[26];
  const unsigned char* mask = (const unsigned char*)d_in[27];
  float* out = (float*)d_out;

  // ---- carve workspace ----
  char* w = (char*)d_ws;
  auto alloc = [&](size_t bytes) {
    char* p = w;
    w += (bytes + 255) & ~(size_t)255;
    return p;
  };
  _Float16* Wih0h = (_Float16*)alloc(768 * 256 * 2);
  _Float16* Whh0h = (_Float16*)alloc(768 * 256 * 2);
  _Float16* Wih1h = (_Float16*)alloc(768 * 256 * 2);
  _Float16* Whh1h = (_Float16*)alloc(768 * 256 * 2);
  _Float16* Wqh   = (_Float16*)alloc(256 * 256 * 2);
  _Float16* Wkh   = (_Float16*)alloc(256 * 256 * 2);
  _Float16* Wvh   = (_Float16*)alloc(256 * 256 * 2);
  _Float16* Woh   = (_Float16*)alloc(256 * 256 * 2);
  _Float16* Wch   = (_Float16*)alloc(256 * 512 * 2);
  _Float16* W1h   = (_Float16*)alloc(256 * 256 * 2);
  _Float16* W2h   = (_Float16*)alloc(256 * 2);
  _Float16* kh    = (_Float16*)alloc((size_t)Bdim * Ndim * 256 * 2);
  _Float16* vth   = (_Float16*)alloc((size_t)Bdim * 256 * Ndim * 2);
  float*    gi    = (float*)alloc((size_t)Bdim * 768 * 4);
  float*    gh    = (float*)alloc((size_t)Bdim * 768 * 4);
  float*    h0f   = (float*)alloc((size_t)Bdim * Hdim * 4);
  float*    h1f   = (float*)alloc((size_t)Bdim * Hdim * 4);
  _Float16* h0h   = (_Float16*)alloc((size_t)Bdim * Hdim * 2);
  _Float16* cath  = (_Float16*)alloc((size_t)Bdim * 512 * 2);  // [h1 | ctx*Wo]
  _Float16* embh  = (_Float16*)alloc((size_t)Bdim * Edim * 2);
  _Float16* qh    = (_Float16*)alloc((size_t)Bdim * Hdim * 2);
  _Float16* ctxh  = (_Float16*)alloc((size_t)Bdim * Hdim * 2);
  _Float16* combh = (_Float16*)alloc((size_t)Bdim * Edim * 2);
  _Float16* out1h = (_Float16*)alloc((size_t)Bdim * Edim * 2);

  // ---- weight conversion f32 -> f16 (L2 resident thereafter) ----
  struct Conv { const float* s; _Float16* d; int n; } convs[] = {
      {Wih0, Wih0h, 768 * 256}, {Whh0, Whh0h, 768 * 256},
      {Wih1, Wih1h, 768 * 256}, {Whh1, Whh1h, 768 * 256},
      {Wq, Wqh, 256 * 256},     {Wk, Wkh, 256 * 256},
      {Wv, Wvh, 256 * 256},     {Wo, Woh, 256 * 256},
      {Wc, Wch, 256 * 512},     {W1, W1h, 256 * 256},
      {W2, W2h, 256}};
  for (auto& c : convs)
    f32_to_f16_kernel<<<(c.n + 255) / 256, 256, 0, stream>>>(c.s, c.d, c.n);

  init_h_kernel<<<(Bdim * Hdim) / 256, 256, 0, stream>>>(h0in, h0f, h0h, h1f, cath);

  // ---- k/v projections: M = B*N = 131072, K = 256, N = 256 ----
  gemm_kernel<float><<<dim3(Bdim * Ndim / 32, 2), 256, 0, stream>>>(
      prev, 256, Wkh, bk, nullptr, 0, kh, 256, Bdim * Ndim, 256, 256, 0, 0);
  gemm_kernel<float><<<dim3(Bdim * Ndim / 32, 2), 256, 0, stream>>>(
      prev, 256, Wvh, bv, nullptr, 0, vth, 0, Bdim * Ndim, 256, 256, 0, 1);

  // ---- sequential scan over T ----
  const dim3 g768(Bdim / 32, 6);
  const dim3 g256(Bdim / 32, 2);
  for (int t = 0; t < Tdim; ++t) {
    emb_step_kernel<<<(Bdim * Edim) / 256, 256, 0, stream>>>(x, W_in, b_in, embh, t);

    // GRU layer 0
    gemm_kernel<_Float16><<<g768, 256, 0, stream>>>(
        embh, 256, Wih0h, nullptr, gi, 768, nullptr, 0, Bdim, 768, 256, 0, 0);
    gemm_kernel<_Float16><<<g768, 256, 0, stream>>>(
        h0h, 256, Whh0h, nullptr, gh, 768, nullptr, 0, Bdim, 768, 256, 0, 0);
    gru_pw_kernel<<<(Bdim * Hdim) / 256, 256, 0, stream>>>(
        gi, gh, bih0, bhh0, h0f, h0f, h0h, 256);

    // GRU layer 1 (f16 state lives in cath[:, 0:256], lda = 512)
    gemm_kernel<_Float16><<<g768, 256, 0, stream>>>(
        h0h, 256, Wih1h, nullptr, gi, 768, nullptr, 0, Bdim, 768, 256, 0, 0);
    gemm_kernel<_Float16><<<g768, 256, 0, stream>>>(
        cath, 512, Whh1h, nullptr, gh, 768, nullptr, 0, Bdim, 768, 256, 0, 0);
    gru_pw_kernel<<<(Bdim * Hdim) / 256, 256, 0, stream>>>(
        gi, gh, bih1, bhh1, h1f, h1f, cath, 512);

    // q = (h1 Wq^T + bq) * 1/sqrt(DH)
    gemm_kernel<_Float16><<<g256, 256, 0, stream>>>(
        cath, 512, Wqh, bq, nullptr, 0, qh, 256, Bdim, 256, 256, 2, 0);

    // attention -> ctx
    attention_kernel<<<(Bdim * HEADS) / 8, 256, 0, stream>>>(qh, kh, vth, mask, ctxh);

    // ctx Wo^T + bo -> cath[:, 256:512]
    gemm_kernel<_Float16><<<g256, 256, 0, stream>>>(
        ctxh, 256, Woh, bo, nullptr, 0, cath + 256, 512, Bdim, 256, 256, 0, 0);

    // comb = relu([h1 | ctxWo] Wc^T + bc), K = 512
    gemm_kernel<_Float16><<<g256, 256, 0, stream>>>(
        cath, 512, Wch, bc, nullptr, 0, combh, 256, Bdim, 256, 512, 1, 0);

    // out1 = relu(comb W1^T + b1)
    gemm_kernel<_Float16><<<g256, 256, 0, stream>>>(
        combh, 256, W1h, b1, nullptr, 0, out1h, 256, Bdim, 256, 256, 1, 0);

    // logits + sigmoid
    final_out_kernel<<<Bdim / 8, 256, 0, stream>>>(out1h, W2h, b2, out, t);
  }
}
